// PolyDecoderHead_55164559950046
// MI455X (gfx1250) — matrix operand order repaired and checked by reading, not compiled, not run on hardware
//
#include <hip/hip_runtime.h>
#include <hip/hip_bf16.h>

#define B_   16
#define C_   256
#define H_   64
#define W_   96
#define HW_  6144
#define Q_   300
#define ROWS_Q (B_*Q_)          // 4800
#define SPROWS (B_*Q_*8)        // 38400

typedef __attribute__((ext_vector_type(16))) __bf16 v16bf;
typedef __attribute__((ext_vector_type(8)))  float  v8f;
typedef __attribute__((ext_vector_type(4)))  unsigned int u32x4;

union FragB { unsigned int u[8]; u32x4 q[2]; v16bf v; };

__device__ __forceinline__ unsigned short f2bf(float f) {
    unsigned int x = __builtin_bit_cast(unsigned int, f);
    x += 0x7fffu + ((x >> 16) & 1u);          // RNE
    return (unsigned short)(x >> 16);
}

__device__ __forceinline__ v8f vzero8() {
    v8f z;
#pragma unroll
    for (int i = 0; i < 8; ++i) z[i] = 0.f;
    return z;
}

__device__ __forceinline__ v8f wmma_bf16(v16bf a, v16bf b, v8f c) {
    return __builtin_amdgcn_wmma_f32_16x16x32_bf16(false, a, false, b, (short)0, c, false, false);
}

// A fragment (16x32 bf16) from LDS row-major [row][k].
// ldk (u16 units) must be a multiple of 8 so each lane's dword groups are 16B aligned
// -> two ds_load_b128 per fragment.
__device__ __forceinline__ v16bf ldsA(const unsigned short* sm, int ldk, int row0, int kt) {
    int lane = threadIdx.x & 31;
    const unsigned int* p = (const unsigned int*)(sm + (size_t)(row0 + (lane & 15)) * ldk)
                            + kt * 16 + ((lane >> 4) << 2);
    FragB f;
    f.q[0] = *(const u32x4*)p;
    f.q[1] = *(const u32x4*)(p + 8);
    return f.v;
}

// B fragment (32x16 bf16) from LDS where B is stored transposed row-major [n][k].
// ldk/2 must be a multiple of 4 -> two ds_load_b128 per fragment.
__device__ __forceinline__ v16bf ldsB_nk(const unsigned short* sm, int ldk, int n0, int kPairBase) {
    int lane = threadIdx.x & 31;
    const unsigned int* p = (const unsigned int*)(sm + (size_t)(n0 + (lane & 15)) * ldk)
                            + kPairBase + ((lane >> 4) << 2);
    FragB f;
    f.q[0] = *(const u32x4*)p;
    f.q[1] = *(const u32x4*)(p + 8);
    return f.v;
}

// B fragment from pre-packed global weights: layout [frag][lane][8 u32]
// -> two fully-coalesced global_load_b128 per fragment.
__device__ __forceinline__ v16bf gB(const unsigned int* pack, int ntiles, int kt, int nt) {
    int lane = threadIdx.x & 31;
    const u32x4* p = (const u32x4*)(pack + (((size_t)(kt * ntiles + nt)) << 8) + (lane << 3));
    FragB f;
    f.q[0] = p[0];
    f.q[1] = p[1];
    return f.v;
}

// ---------------- x [B,C,H,W] -> xt [B,HW,C] (NHWC) ----------------
__global__ void k_transpose(const float* __restrict__ x, float* __restrict__ xt) {
    __shared__ float t[32][33];
    int b = blockIdx.z;
    int p0 = blockIdx.x * 32, c0 = blockIdx.y * 32;
#pragma unroll
    for (int i = 0; i < 4; ++i) {
        int c = c0 + threadIdx.y + i * 8;
        t[threadIdx.y + i * 8][threadIdx.x] = x[((size_t)b * C_ + c) * HW_ + p0 + threadIdx.x];
    }
    __syncthreads();
#pragma unroll
    for (int i = 0; i < 4; ++i) {
        int p = p0 + threadIdx.y + i * 8;
        xt[((size_t)b * HW_ + p) * C_ + c0 + threadIdx.x] = t[threadIdx.x][threadIdx.y + i * 8];
    }
}

// ---------------- collapse rp_w1@rp_w2 (no nonlinearity between them) ----------------
__global__ void k_rp_collapse(const float* __restrict__ w1, const float* __restrict__ b1,
                              const float* __restrict__ w2, const float* __restrict__ b2,
                              float* __restrict__ weff, float* __restrict__ beff) {
    int t = blockIdx.x * blockDim.x + threadIdx.x;
    if (t < 256 * 16) {
        int c = t >> 4, j = t & 15;
        float s = 0.f;
        for (int k = 0; k < 512; ++k) s += w1[c * 512 + k] * w2[k * 16 + j];
        weff[t] = s;
    } else if (t < 256 * 16 + 16) {
        int j = t - 4096;
        float s = b2[j];
        for (int k = 0; k < 512; ++k) s += b1[k] * w2[k * 16 + j];
        beff[j] = s;
    }
}

// ---------------- pack f32 weight [K][N] into WMMA B-fragment order (bf16) ----------------
// out[t]: t = frag*256 + lane*8 + j  (lane-contiguous for b128 loads)
__global__ void k_pack(const float* __restrict__ w, int K, int N, unsigned int* __restrict__ out) {
    int total = (K / 2) * N;
    for (int t = blockIdx.x * blockDim.x + threadIdx.x; t < total; t += gridDim.x * blockDim.x) {
        int j = t & 7, lane = (t >> 3) & 31, frag = t >> 8;
        int NT = N >> 4;
        int kt = frag / NT, nt = frag % NT;
        int n = nt * 16 + (lane & 15);
        int k0 = kt * 32 + ((lane >> 4) << 3) + ((j >> 2) << 4) + 2 * (j & 3);
        unsigned int lo = f2bf(w[(size_t)k0 * N + n]);
        unsigned int hi = f2bf(w[(size_t)(k0 + 1) * N + n]);
        out[t] = lo | (hi << 16);
    }
}

// ---------------- fused token kernel: qs MLP -> logits, collapsed rp -> tanh ----------------
__global__ void k_score_rp(const float* __restrict__ xt,
                           const unsigned int* __restrict__ qsP, const float* __restrict__ qs_b1,
                           const float* __restrict__ qs_w2, const float* __restrict__ qs_b2,
                           const unsigned int* __restrict__ rpP, const float* __restrict__ rp_beff,
                           float* __restrict__ logits, float* __restrict__ rp_out) {
    extern __shared__ unsigned short sm[];           // [128][264] bf16
    const int LDK = 264;
    int rowbase = blockIdx.x * 128;
    for (int idx = threadIdx.x; idx < 128 * 256; idx += 256) {
        int c = idx & 255, r = idx >> 8;
        sm[r * LDK + c] = f2bf(xt[(size_t)(rowbase + r) * 256 + c]);
    }
    __builtin_prefetch(qsP + (threadIdx.x << 5), 0, 1);
    __syncthreads();
    int wave = threadIdx.x >> 5, lane = threadIdx.x & 31;
    int r0 = wave * 16;
    v16bf afr[8];
#pragma unroll
    for (int kt = 0; kt < 8; ++kt) afr[kt] = ldsA(sm, LDK, r0, kt);

    float lpart[8];
#pragma unroll
    for (int r = 0; r < 8; ++r) lpart[r] = 0.f;
    for (int nt = 0; nt < 32; ++nt) {
        v8f acc = vzero8();
#pragma unroll
        for (int kt = 0; kt < 8; ++kt) acc = wmma_bf16(afr[kt], gB(qsP, 32, kt, nt), acc);
        int ncol = nt * 16 + (lane & 15);
        float bb = qs_b1[ncol], w2 = qs_w2[ncol];
#pragma unroll
        for (int r = 0; r < 8; ++r) {
            float h = acc[r] + bb;
            h = h > 0.f ? h : (__expf(h) - 1.f);       // ELU
            lpart[r] += h * w2;
        }
    }
#pragma unroll
    for (int r = 0; r < 8; ++r) {
        float v = lpart[r];
        for (int off = 1; off < 16; off <<= 1) v += __shfl_xor(v, off, 32);
        if ((lane & 15) == 0) {
            int row = rowbase + r0 + r + ((lane >> 4) << 3);
            logits[row] = v + qs_b2[0];
        }
    }
    {   // collapsed rp branch: N=16
        v8f acc = vzero8();
#pragma unroll
        for (int kt = 0; kt < 8; ++kt) acc = wmma_bf16(afr[kt], gB(rpP, 1, kt, 0), acc);
        int j = lane & 15;
        float bb = rp_beff[j];
#pragma unroll
        for (int r = 0; r < 8; ++r) {
            int row = rowbase + r0 + r + ((lane >> 4) << 3);
            rp_out[(size_t)row * 16 + j] = tanhf(acc[r] + bb);
        }
    }
}

// ---------------- top-k (300 of 6144) per batch, iterative argmax ----------------
__global__ void k_topk(const float* __restrict__ logits, int* __restrict__ idxs) {
    __shared__ float vals[HW_];
    __shared__ float bv[256];
    __shared__ int   bi[256];
    int b = blockIdx.x, t = threadIdx.x;
    for (int i = t; i < HW_; i += 256) vals[i] = logits[b * HW_ + i];
    __syncthreads();
    for (int s = 0; s < Q_; ++s) {
        float best = -3.0e38f; int bidx = 0;
        for (int i = t; i < HW_; i += 256) {
            float v = vals[i];
            if (v > best) { best = v; bidx = i; }
        }
        bv[t] = best; bi[t] = bidx;
        __syncthreads();
        for (int off = 128; off > 0; off >>= 1) {
            if (t < off) {
                float v2 = bv[t + off]; int i2 = bi[t + off];
                if (v2 > bv[t] || (v2 == bv[t] && i2 < bi[t])) { bv[t] = v2; bi[t] = i2; }
            }
            __syncthreads();
        }
        if (t == 0) { idxs[b * Q_ + s] = bi[0]; vals[bi[0]] = -3.0e38f; }
        __syncthreads();
    }
}

// ---------------- gather querys and reference points ----------------
__global__ void k_gather(const float* __restrict__ xt, const float* __restrict__ rp,
                         const int* __restrict__ idxs, float* __restrict__ querys,
                         float* __restrict__ refb) {
    int bq = blockIdx.x; int b = bq / Q_; int c = threadIdx.x;
    int p = idxs[bq];
    querys[(size_t)bq * 256 + c] = xt[((size_t)b * HW_ + p) * 256 + c];
    if (c < 16) refb[bq * 16 + c] = rp[(size_t)(b * HW_ + p) * 16 + c];
}

// ---------------- generic WMMA GEMM: Out = act(A[rows,256] @ W + bias) (+resid) ----------------
__global__ void k_gemm256(const float* __restrict__ A, int rows,
                          const unsigned int* __restrict__ Wp, const float* __restrict__ bias,
                          int N, float* __restrict__ Out, int act,
                          const float* __restrict__ resid) {
    extern __shared__ unsigned short sm[];           // [128][264] bf16
    const int LDK = 264;
    int rowbase = blockIdx.x * 128;
    for (int idx = threadIdx.x; idx < 128 * 256; idx += 256) {
        int c = idx & 255, r = idx >> 8;
        int row = rowbase + r;
        float v = (row < rows) ? A[(size_t)row * 256 + c] : 0.f;
        sm[r * LDK + c] = f2bf(v);
    }
    __builtin_prefetch(Wp + (threadIdx.x << 5), 0, 1);
    __syncthreads();
    int wave = threadIdx.x >> 5, lane = threadIdx.x & 31;
    int r0 = wave * 16;
    v16bf afr[8];
#pragma unroll
    for (int kt = 0; kt < 8; ++kt) afr[kt] = ldsA(sm, LDK, r0, kt);
    int NT = N >> 4;
    for (int nt = 0; nt < NT; ++nt) {
        v8f acc = vzero8();
#pragma unroll
        for (int kt = 0; kt < 8; ++kt) acc = wmma_bf16(afr[kt], gB(Wp, NT, kt, nt), acc);
        int col = nt * 16 + (lane & 15);
        float bb = bias ? bias[col] : 0.f;
#pragma unroll
        for (int r = 0; r < 8; ++r) {
            int row = rowbase + r0 + r + ((lane >> 4) << 3);
            if (row < rows) {
                float v = acc[r] + bb;
                if (act == 1)      v = v / (1.f + __expf(-v));              // silu
                else if (act == 2) v = v > 0.f ? v : (__expf(v) - 1.f);     // elu
                if (resid) v += resid[(size_t)row * N + col];
                Out[(size_t)row * N + col] = v;
            }
        }
    }
}

// ---------------- MHA core per (b,h): WMMA scores, softmax, WMMA P@V ----------------
__global__ void k_attn(const float* __restrict__ qkv, float* __restrict__ obuf) {
    extern __shared__ char dyn[];
    unsigned short* kS = (unsigned short*)dyn;                    // [304][40] bf16, K stored [seq][d]
    unsigned short* vT = kS + 304 * 40;                           // [32][328] bf16, V transposed [d][seq]
    float* Sbase = (float*)(vT + 32 * 328);                       // per-wave [16][320] f32
    unsigned short* Pbase = (unsigned short*)(Sbase + 4 * 16 * 320); // per-wave [16][328] bf16
    int b = blockIdx.x, h = blockIdx.y;
    int wave = threadIdx.x >> 5, lane = threadIdx.x & 31;
    const float scale = 0.17677669529663687f;                     // 1/sqrt(32)

    for (int idx = threadIdx.x; idx < 304 * 32; idx += 128) {
        int d = idx & 31, r = idx >> 5;
        float v = (r < Q_) ? qkv[((size_t)(b * Q_ + r)) * 768 + 256 + h * 32 + d] : 0.f;
        kS[r * 40 + d] = f2bf(v);
    }
    for (int idx = threadIdx.x; idx < 320 * 32; idx += 128) {
        int d = idx & 31, r = idx >> 5;
        float v = (r < Q_) ? qkv[((size_t)(b * Q_ + r)) * 768 + 512 + h * 32 + d] : 0.f;
        vT[d * 328 + r] = f2bf(v);
    }
    __syncthreads();

    float* S = Sbase + wave * 16 * 320;
    unsigned short* P = Pbase + wave * 16 * 328;

    for (int qt = wave; qt < 19; qt += 4) {
        FragB qa;                                   // q A-fragment built straight from global
        {
            int row = qt * 16 + (lane & 15);
            if (row >= Q_) row = 0;                 // padded rows: outputs never stored
            const float* qp = qkv + ((size_t)(b * Q_ + row)) * 768 + h * 32;
            int hi = lane >> 4;
#pragma unroll
            for (int j = 0; j < 4; ++j) {
                int kk0 = hi * 8 + 2 * j;
                int kk1 = 16 + hi * 8 + 2 * j;
                qa.u[j]     = (unsigned)f2bf(qp[kk0]) | ((unsigned)f2bf(qp[kk0 + 1]) << 16);
                qa.u[4 + j] = (unsigned)f2bf(qp[kk1]) | ((unsigned)f2bf(qp[kk1 + 1]) << 16);
            }
        }
        for (int kt2 = 0; kt2 < 19; ++kt2) {        // scores = q @ k^T
            v8f acc = vzero8();
            acc = wmma_bf16(qa.v, ldsB_nk(kS, 40, kt2 * 16, 0), acc);
            int ncol = kt2 * 16 + (lane & 15);
#pragma unroll
            for (int r = 0; r < 8; ++r)
                S[(r + ((lane >> 4) << 3)) * 320 + ncol] = acc[r] * scale;
        }
        for (int mrow = 0; mrow < 16; ++mrow) {     // softmax over valid 300 cols
            float* Sr = S + mrow * 320;
            float m = -3.0e38f;
            for (int i = lane; i < Q_; i += 32) m = fmaxf(m, Sr[i]);
            for (int off = 16; off > 0; off >>= 1) m = fmaxf(m, __shfl_xor(m, off, 32));
            float ssum = 0.f;
            for (int i = lane; i < Q_; i += 32) { float e = __expf(Sr[i] - m); Sr[i] = e; ssum += e; }
            for (int off = 16; off > 0; off >>= 1) ssum += __shfl_xor(ssum, off, 32);
            float inv = 1.f / ssum;
            unsigned short* Pr = P + mrow * 328;
            for (int i = lane; i < Q_; i += 32) Pr[i] = f2bf(Sr[i] * inv);
            for (int i = Q_ + lane; i < 320; i += 32) Pr[i] = 0;
        }
#pragma unroll
        for (int nt = 0; nt < 2; ++nt) {            // O = P @ V
            v8f acc = vzero8();
            for (int kt2 = 0; kt2 < 10; ++kt2)
                acc = wmma_bf16(ldsA(P, 328, 0, kt2), ldsB_nk(vT, 328, nt * 16, kt2 * 16), acc);
            int d = nt * 16 + (lane & 15);
#pragma unroll
            for (int r = 0; r < 8; ++r) {
                int qrow = qt * 16 + r + ((lane >> 4) << 3);
                if (qrow < Q_)
                    obuf[((size_t)(b * Q_ + qrow)) * 256 + h * 32 + d] = acc[r];
            }
        }
    }
}

// ---------------- LayerNorm per row of 256 ----------------
__global__ void k_ln(const float* __restrict__ in, const float* __restrict__ g,
                     const float* __restrict__ bb, float* __restrict__ out) {
    __shared__ float red[256];
    int row = blockIdx.x, t = threadIdx.x;
    float v = in[(size_t)row * 256 + t];
    red[t] = v; __syncthreads();
    for (int off = 128; off > 0; off >>= 1) { if (t < off) red[t] += red[t + off]; __syncthreads(); }
    float mu = red[0] * (1.f / 256.f); __syncthreads();
    float d = v - mu;
    red[t] = d * d; __syncthreads();
    for (int off = 128; off > 0; off >>= 1) { if (t < off) red[t] += red[t + off]; __syncthreads(); }
    float var = red[0] * (1.f / 256.f);
    out[(size_t)row * 256 + t] = d * rsqrtf(var + 1e-5f) * g[t] + bb[t];
}

// ---------------- bilinear grid sample, writes sp in [B, n*300+q, C] order ----------------
__global__ void k_sample(const float* __restrict__ xt, const float* __restrict__ refb,
                         float* __restrict__ sp) {
    int bq = blockIdx.x; int b = bq / Q_; int q = bq % Q_;
    int c = threadIdx.x;
#pragma unroll
    for (int n = 0; n < 8; ++n) {
        float gx = refb[bq * 16 + 2 * n];
        float gy = refb[bq * 16 + 2 * n + 1];
        float ix = (gx + 1.f) * (W_ * 0.5f) - 0.5f;
        float iy = (gy + 1.f) * (H_ * 0.5f) - 0.5f;
        float fx = floorf(ix), fy = floorf(iy);
        int x0 = (int)fx, y0 = (int)fy;
        float wx1 = ix - fx, wy1 = iy - fy;
        float wx0 = 1.f - wx1, wy0 = 1.f - wy1;
        auto samp = [&](int yy, int xx) -> float {
            if (xx < 0 || xx >= W_ || yy < 0 || yy >= H_) return 0.f;
            return xt[(((size_t)b * HW_) + yy * W_ + xx) * 256 + c];
        };
        float acc = samp(y0, x0)         * wy0 * wx0
                  + samp(y0, x0 + 1)     * wy0 * wx1
                  + samp(y0 + 1, x0)     * wy1 * wx0
                  + samp(y0 + 1, x0 + 1) * wy1 * wx1;
        sp[((size_t)b * 2400 + n * Q_ + q) * 256 + c] = acc;
    }
}

// ---------------- deformable attention over 8 points (faithful row-order mismatch) ----------------
__global__ void k_deform(const float* __restrict__ dq, const float* __restrict__ dk,
                         const float* __restrict__ dv, float* __restrict__ tgt) {
    int bq = blockIdx.x; int b = bq / Q_, q = bq % Q_;
    int t = threadIdx.x; int h = t >> 5, lane = t & 31;
    const float scale = 0.17677669529663687f;
    float qv = dq[(size_t)bq * 256 + h * 32 + lane];
    float sc[8];
#pragma unroll
    for (int o = 0; o < 8; ++o) {
        size_t row = (size_t)b * 2400 + q * 8 + o;      // reshape-faithful indexing
        float p = qv * dk[row * 256 + h * 32 + lane];
        for (int off = 16; off > 0; off >>= 1) p += __shfl_xor(p, off, 32);
        sc[o] = p * scale;
    }
    float m = sc[0];
#pragma unroll
    for (int o = 1; o < 8; ++o) m = fmaxf(m, sc[o]);
    float s = 0.f;
#pragma unroll
    for (int o = 0; o < 8; ++o) { sc[o] = __expf(sc[o] - m); s += sc[o]; }
    float inv = 1.f / s;
    float acc = 0.f;
#pragma unroll
    for (int o = 0; o < 8; ++o) {
        size_t row = (size_t)b * 2400 + q * 8 + o;
        acc += sc[o] * inv * dv[row * 256 + h * 32 + lane];
    }
    tgt[(size_t)bq * 256 + t] += acc;
}

// ---------------- fused poly / cls output heads ----------------
__global__ void k_polycls(const float* __restrict__ tgt,
                          const unsigned int* __restrict__ poP, const float* __restrict__ po_b1,
                          const float* __restrict__ po_w2, const float* __restrict__ po_b2,
                          const unsigned int* __restrict__ clP, const float* __restrict__ cl_b1,
                          const float* __restrict__ cl_w2, const float* __restrict__ cl_b2,
                          float* __restrict__ out) {
    extern __shared__ unsigned short sm[];           // [128][264] bf16
    const int LDK = 264;
    const int rows = ROWS_Q;
    int rowbase = blockIdx.x * 128;
    for (int idx = threadIdx.x; idx < 128 * 256; idx += 256) {
        int c = idx & 255, r = idx >> 8;
        int row = rowbase + r;
        sm[r * LDK + c] = f2bf(row < rows ? tgt[(size_t)row * 256 + c] : 0.f);
    }
    __syncthreads();
    int wave = threadIdx.x >> 5, lane = threadIdx.x & 31;
    int r0 = wave * 16;
    v16bf afr[8];
#pragma unroll
    for (int kt = 0; kt < 8; ++kt) afr[kt] = ldsA(sm, LDK, r0, kt);

    float pp[8][8], cp[8];
#pragma unroll
    for (int r = 0; r < 8; ++r) { cp[r] = 0.f; for (int j = 0; j < 8; ++j) pp[r][j] = 0.f; }

    for (int nt = 0; nt < 32; ++nt) {                // poly: silu hidden + 512x8 matvec
        v8f acc = vzero8();
#pragma unroll
        for (int kt = 0; kt < 8; ++kt) acc = wmma_bf16(afr[kt], gB(poP, 32, kt, nt), acc);
        int ncol = nt * 16 + (lane & 15);
        float bb = po_b1[ncol];
        float w2[8];
#pragma unroll
        for (int j = 0; j < 8; ++j) w2[j] = po_w2[ncol * 8 + j];
#pragma unroll
        for (int r = 0; r < 8; ++r) {
            float h = acc[r] + bb;
            h = h / (1.f + __expf(-h));
#pragma unroll
            for (int j = 0; j < 8; ++j) pp[r][j] += h * w2[j];
        }
    }
    for (int nt = 0; nt < 32; ++nt) {                // cls: elu hidden + 512x1 matvec
        v8f acc = vzero8();
#pragma unroll
        for (int kt = 0; kt < 8; ++kt) acc = wmma_bf16(afr[kt], gB(clP, 32, kt, nt), acc);
        int ncol = nt * 16 + (lane & 15);
        float bb = cl_b1[ncol], w2 = cl_w2[ncol];
#pragma unroll
        for (int r = 0; r < 8; ++r) {
            float h = acc[r] + bb;
            h = h > 0.f ? h : (__expf(h) - 1.f);
            cp[r] += h * w2;
        }
    }
#pragma unroll
    for (int r = 0; r < 8; ++r) {
#pragma unroll
        for (int j = 0; j < 8; ++j) {
            float v = pp[r][j];
            for (int off = 1; off < 16; off <<= 1) v += __shfl_xor(v, off, 32);
            pp[r][j] = v;
        }
        float cv = cp[r];
        for (int off = 1; off < 16; off <<= 1) cv += __shfl_xor(cv, off, 32);
        if ((lane & 15) == 0) {
            int row = rowbase + r0 + r + ((lane >> 4) << 3);
            if (row < rows) {
#pragma unroll
                for (int j = 0; j < 8; ++j) out[(size_t)row * 8 + j] = pp[r][j] + po_b2[j];
                out[38400 + row] = 1.f / (1.f + __expf(-(cv + cl_b2[0])));
            }
        }
    }
}

extern "C" void kernel_launch(void* const* d_in, const int* in_sizes, int n_in,
                              void* d_out, int out_size, void* d_ws, size_t ws_size,
                              hipStream_t stream) {
    const float* x        = (const float*)d_in[0];
    const float* qs_w1    = (const float*)d_in[1];
    const float* qs_b1    = (const float*)d_in[2];
    const float* qs_w2    = (const float*)d_in[3];
    const float* qs_b2    = (const float*)d_in[4];
    const float* rp_w1    = (const float*)d_in[5];
    const float* rp_b1    = (const float*)d_in[6];
    const float* rp_w2    = (const float*)d_in[7];
    const float* rp_b2    = (const float*)d_in[8];
    const float* mha_in_w = (const float*)d_in[9];
    const float* mha_in_b = (const float*)d_in[10];
    const float* mha_out_w= (const float*)d_in[11];
    const float* mha_out_b= (const float*)d_in[12];
    const float* dq_w     = (const float*)d_in[13];
    const float* dq_b     = (const float*)d_in[14];
    const float* dk_w     = (const float*)d_in[15];
    const float* dk_b     = (const float*)d_in[16];
    const float* dv_w     = (const float*)d_in[17];
    const float* dv_b     = (const float*)d_in[18];
    const float* ln1_g    = (const float*)d_in[19];
    const float* ln1_b    = (const float*)d_in[20];
    const float* cl_w1    = (const float*)d_in[21];
    const float* cl_b1    = (const float*)d_in[22];
    const float* cl_w2    = (const float*)d_in[23];
    const float* cl_b2    = (const float*)d_in[24];
    const float* po_w1    = (const float*)d_in[25];
    const float* po_b1    = (const float*)d_in[26];
    const float* po_w2    = (const float*)d_in[27];
    const float* po_b2    = (const float*)d_in[28];
    float* out = (float*)d_out;

    float* ws = (float*)d_ws;
    size_t o = 0;
    float* xt     = ws + o; o += (size_t)B_ * HW_ * C_;      // 25,165,824
    float* rp_buf = ws + o; o += (size_t)B_ * HW_ * 16;      //  1,572,864
    float* logits = ws + o; o += (size_t)B_ * HW_;           //     98,304
    int*   idxs   = (int*)(ws + o); o += ROWS_Q;             //      4,800
    float* querys = ws + o; o += (size_t)ROWS_Q * 256;       //  1,228,800
    float* refb   = ws + o; o += (size_t)ROWS_Q * 16;        //     76,800
    float* qkv    = ws + o; o += (size_t)ROWS_Q * 768;       //  3,686,400
    float* obuf   = ws + o; o += (size_t)ROWS_Q * 256;
    float* tgt    = ws + o; o += (size_t)ROWS_Q * 256;
    float* dqb    = ws + o; o += (size_t)ROWS_Q * 256;
    float* sp     = ws + o; o += (size_t)SPROWS * 256;       //  9,830,400
    float* weff   = ws + o; o += 256 * 16;
    float* beff   = ws + o; o += 16;
    unsigned int* qsP  = (unsigned int*)(ws + o); o += 256 * 512 / 2;
    unsigned int* rpP  = (unsigned int*)(ws + o); o += 256 * 16 / 2;
    unsigned int* inP  = (unsigned int*)(ws + o); o += 256 * 768 / 2;
    unsigned int* outP = (unsigned int*)(ws + o); o += 256 * 256 / 2;
    unsigned int* dqP  = (unsigned int*)(ws + o); o += 256 * 256 / 2;
    unsigned int* dkP  = (unsigned int*)(ws + o); o += 256 * 256 / 2;
    unsigned int* dvP  = (unsigned int*)(ws + o); o += 256 * 256 / 2;
    unsigned int* poP  = (unsigned int*)(ws + o); o += 256 * 512 / 2;
    unsigned int* clP  = (unsigned int*)(ws + o); o += 256 * 512 / 2;
    // xt is dead after k_gather/k_sample -> alias dk/dv buffers into it (19.7M < 25.2M floats)
    float* dkb = xt;
    float* dvb = xt + (size_t)SPROWS * 256;

    const size_t ldsGemm = 128 * 264 * sizeof(unsigned short);              // 67,584 B
    const size_t ldsAttn = 304 * 40 * 2 + 32 * 328 * 2
                         + 4 * 16 * 320 * 4 + 4 * 16 * 328 * 2;             // 169,216 B

    // 1) NHWC transpose
    k_transpose<<<dim3(HW_ / 32, C_ / 32, B_), dim3(32, 8), 0, stream>>>(x, xt);
    // 2) collapse rp MLP
    k_rp_collapse<<<17, 256, 0, stream>>>(rp_w1, rp_b1, rp_w2, rp_b2, weff, beff);
    // 3) pack weights into WMMA B-fragment order
    auto pack = [&](const float* w, int K, int N, unsigned int* dst) {
        int total = (K / 2) * N;
        k_pack<<<(total + 255) / 256, 256, 0, stream>>>(w, K, N, dst);
    };
    pack(qs_w1, 256, 512, qsP);
    pack(weff, 256, 16, rpP);
    pack(mha_in_w, 256, 768, inP);
    pack(mha_out_w, 256, 256, outP);
    pack(dq_w, 256, 256, dqP);
    pack(dk_w, 256, 256, dkP);
    pack(dv_w, 256, 256, dvP);
    pack(po_w1, 256, 512, poP);
    pack(cl_w1, 256, 512, clP);
    // 4) fused token kernel (heaviest GEMM on the chip)
    k_score_rp<<<(B_ * HW_) / 128, 256, ldsGemm, stream>>>(xt, qsP, qs_b1, qs_w2, qs_b2,
                                                           rpP, beff, logits, rp_buf);
    // 5) top-k
    k_topk<<<B_, 256, 0, stream>>>(logits, idxs);
    // 6) gather
    k_gather<<<ROWS_Q, 256, 0, stream>>>(xt, rp_buf, idxs, querys, refb);
    // 7) qkv projection
    k_gemm256<<<(ROWS_Q + 127) / 128, 256, ldsGemm, stream>>>(querys, ROWS_Q, inP, mha_in_b,
                                                              768, qkv, 0, nullptr);
    // 8) attention
    k_attn<<<dim3(B_, 8), 128, ldsAttn, stream>>>(qkv, obuf);
    // 9) out projection + residual, then LayerNorm in place
    k_gemm256<<<(ROWS_Q + 127) / 128, 256, ldsGemm, stream>>>(obuf, ROWS_Q, outP, mha_out_b,
                                                              256, tgt, 0, querys);
    k_ln<<<ROWS_Q, 256, 0, stream>>>(tgt, ln1_g, ln1_b, tgt);
    // 10) dq projection
    k_gemm256<<<(ROWS_Q + 127) / 128, 256, ldsGemm, stream>>>(tgt, ROWS_Q, dqP, dq_b,
                                                              256, dqb, 0, nullptr);
    // 11) grid sample (last consumer of xt)
    k_sample<<<ROWS_Q, 256, 0, stream>>>(xt, refb, sp);
    // 12) dk/dv projections (write into xt alias)
    k_gemm256<<<(SPROWS + 127) / 128, 256, ldsGemm, stream>>>(sp, SPROWS, dkP, dk_b,
                                                              256, dkb, 0, nullptr);
    k_gemm256<<<(SPROWS + 127) / 128, 256, ldsGemm, stream>>>(sp, SPROWS, dvP, dv_b,
                                                              256, dvb, 0, nullptr);
    // 13) deformable attention + residual (tgt updated in place)
    k_deform<<<ROWS_Q, 256, 0, stream>>>(dqb, dkb, dvb, tgt);
    // 14) output heads -> d_out (poly[0..38399], cls[38400..43199])
    k_polycls<<<(ROWS_Q + 127) / 128, 256, ldsGemm, stream>>>(tgt, poP, po_b1, po_w2, po_b2,
                                                              clP, cl_b1, cl_w2, cl_b2, out);
    (void)in_sizes; (void)n_in; (void)out_size; (void)ws_size;
}
